// ImgGRU_6871947674137
// MI455X (gfx1250) — compile-verified, matrix operand order checked by reading, not run on hardware
//
#include <hip/hip_runtime.h>
#include <hip/hip_bf16.h>

typedef __attribute__((ext_vector_type(16))) _Float16 v16h;
typedef __attribute__((ext_vector_type(8)))  _Float16 v8h;
typedef __attribute__((ext_vector_type(8)))  float    v8f;

#define BATCH 256
#define CH    3
#define HT    224
#define WD    224
#define IN    672     /* C*WD */
#define HID   512
#define TH3   1536    /* 3*HID */
#define DM    512

// ---------------------------------------------------------------------------
// Fragment load: row-major f16 row pointer (already at +k0). CDNA5 16-bit A/B
// layout: lane L holds row/col (L&15); halves 0..7 = K kbase..kbase+7,
// halves 8..15 = K 16+kbase..16+kbase+7, kbase = (L>>4)*8.  Two 16B loads.
// ---------------------------------------------------------------------------
__device__ __forceinline__ v16h load_frag(const _Float16* __restrict__ p, int kbase) {
    v8h lo = *(const v8h*)(p + kbase);
    v8h hi = *(const v8h*)(p + kbase + 16);
    v16h f;
#pragma unroll
    for (int i = 0; i < 8; ++i) { f[i] = lo[i]; f[i + 8] = hi[i]; }
    return f;
}

// ---------------------------------------------------------------------------
// Generic WMMA GEMM (used for the big parallel gx0 GEMM and the head):
// C[M,N] = A[M,K] * B[N,K]^T + bias[N].  Block = 128 thr (4 waves), each wave
// a 64(M) x 16(N) strip, B fragment reused 4x.  grid = (N/64, M/64).
// ---------------------------------------------------------------------------
template<bool OUT_F16, bool RELU>
__global__ __launch_bounds__(128)
void gemm_wmma_kernel(const _Float16* __restrict__ A,
                      const _Float16* __restrict__ B,
                      const float* __restrict__ bias,
                      void* __restrict__ Cout,
                      int M, int N, int K) {
    const int lane  = threadIdx.x & 31;
    const int wave  = threadIdx.x >> 5;
    const int n0    = (blockIdx.x * 4 + wave) * 16;
    const int m0    = blockIdx.y * 64;
    const int hrow  = lane & 15;
    const int kbase = (lane >> 4) * 8;

    const _Float16* __restrict__ Brow = B + (size_t)(n0 + hrow) * K;
    const _Float16* __restrict__ Ar0  = A + (size_t)(m0 + hrow) * K;

    v8f acc0 = {}, acc1 = {}, acc2 = {}, acc3 = {};
    for (int k0 = 0; k0 < K; k0 += 32) {
        v16h bf = load_frag(Brow + k0, kbase);
        v16h a0 = load_frag(Ar0 + k0,                  kbase);
        v16h a1 = load_frag(Ar0 + (size_t)16 * K + k0, kbase);
        v16h a2 = load_frag(Ar0 + (size_t)32 * K + k0, kbase);
        v16h a3 = load_frag(Ar0 + (size_t)48 * K + k0, kbase);
        acc0 = __builtin_amdgcn_wmma_f32_16x16x32_f16(false, a0, false, bf, (short)0, acc0, false, false);
        acc1 = __builtin_amdgcn_wmma_f32_16x16x32_f16(false, a1, false, bf, (short)0, acc1, false, false);
        acc2 = __builtin_amdgcn_wmma_f32_16x16x32_f16(false, a2, false, bf, (short)0, acc2, false, false);
        acc3 = __builtin_amdgcn_wmma_f32_16x16x32_f16(false, a3, false, bf, (short)0, acc3, false, false);
    }

    const int   col  = n0 + hrow;
    const float bv   = bias[col];
    const int   roff = (lane >> 4) * 8;
    float*     Cf = (float*)Cout;
    _Float16*  Ch = (_Float16*)Cout;
    v8f acc[4] = {acc0, acc1, acc2, acc3};
#pragma unroll
    for (int i = 0; i < 4; ++i) {
#pragma unroll
        for (int r = 0; r < 8; ++r) {
            int   row = m0 + 16 * i + roff + r;
            float v   = acc[i][r] + bv;
            if (RELU) v = v > 0.f ? v : 0.f;
            if (OUT_F16) Ch[(size_t)row * N + col] = (_Float16)v;
            else         Cf[(size_t)row * N + col] = v;
        }
    }
}

// ---------------------------------------------------------------------------
// Fused GRU layer stage (one launch per layer per timestep).
// Block = 192 thr (6 waves).  Block tile: rows [m0,m0+64) x gate cols
// [n0,n0+16).  Wave w: combo = (src s = w/3: 0=gh, 1=gi) x (gate g = w%3:
// r,z,n).  Each GEMM wave computes its 64x16 tile of
//   gh = hOld * Whh^T + bhh     (cols g*512+n0 .. +16)
//   gi = xin  * Wih^T + bih
// into LDS; layer 0's gi waves instead copy the precomputed gx tile (f16,
// bias already folded in) to LDS.  After barrier: GRU gate math in-register,
// writing h_new f32 (recurrence) + f16 (next stage's WMMA A operand).
// h is double-buffered across timesteps (hOld != hNew) to avoid cross-block
// RAW races.  grid = (HID/16 = 32, BATCH/64 = 4).
// ---------------------------------------------------------------------------
template<bool L0>
__global__ __launch_bounds__(192)
void gru_stage_kernel(const _Float16* __restrict__ hOldH,  // [B,HID] f16 (t-1, layer l)
                      const float*    __restrict__ hOldF,  // [B,HID] f32
                      const _Float16* __restrict__ xinH,   // [B,HID] f16 layer input (l>0)
                      const _Float16* __restrict__ gx,     // [B,TH3] f16 (L0 only, t-offset applied)
                      const _Float16* __restrict__ Wih,    // [TH3,HID] f16 (l>0)
                      const _Float16* __restrict__ Whh,    // [TH3,HID] f16
                      const float* __restrict__ bih,       // [TH3] (l>0)
                      const float* __restrict__ bhh,       // [TH3]
                      float* __restrict__ hNewF, _Float16* __restrict__ hNewH) {
    __shared__ float smem[6][64][16];                      // 24 KB

    const int tid   = threadIdx.x;
    const int lane  = tid & 31;
    const int wave  = tid >> 5;        // 0..5
    const int g     = wave % 3;        // gate: 0=r 1=z 2=n
    const int s     = wave / 3;        // 0=gh  1=gi
    const int combo = wave;
    const int n0    = blockIdx.x * 16; // gate-column tile within [0,512)
    const int m0    = blockIdx.y * 64; // batch-row tile
    const int hrow  = lane & 15;
    const int kbase = (lane >> 4) * 8;

    if (L0 && s == 1) {
        // gi tile comes precomputed: gx[m0+rr][g*512 + n0 + 0..15], f16.
        const _Float16* src = gx + (size_t)m0 * TH3 + (size_t)g * HID + n0;
        for (int rr = lane; rr < 64; rr += 32) {
            const _Float16* p = src + (size_t)rr * TH3;
            v8h lo = *(const v8h*)p;
            v8h hi = *(const v8h*)(p + 8);
#pragma unroll
            for (int c = 0; c < 8; ++c) {
                smem[combo][rr][c]     = (float)lo[c];
                smem[combo][rr][c + 8] = (float)hi[c];
            }
        }
    } else {
        const _Float16* Amat = (s == 0) ? hOldH : xinH;
        const _Float16* Wmat = (s == 0) ? Whh   : Wih;
        const float*    bias = (s == 0) ? bhh   : bih;

        const _Float16* __restrict__ Brow = Wmat + (size_t)(g * HID + n0 + hrow) * HID;
        const _Float16* __restrict__ Ar0  = Amat + (size_t)(m0 + hrow) * HID;

        v8f acc0 = {}, acc1 = {}, acc2 = {}, acc3 = {};
        for (int k0 = 0; k0 < HID; k0 += 32) {
            v16h bf = load_frag(Brow + k0, kbase);
            v16h a0 = load_frag(Ar0 + k0,                    kbase);
            v16h a1 = load_frag(Ar0 + (size_t)16 * HID + k0, kbase);
            v16h a2 = load_frag(Ar0 + (size_t)32 * HID + k0, kbase);
            v16h a3 = load_frag(Ar0 + (size_t)48 * HID + k0, kbase);
            acc0 = __builtin_amdgcn_wmma_f32_16x16x32_f16(false, a0, false, bf, (short)0, acc0, false, false);
            acc1 = __builtin_amdgcn_wmma_f32_16x16x32_f16(false, a1, false, bf, (short)0, acc1, false, false);
            acc2 = __builtin_amdgcn_wmma_f32_16x16x32_f16(false, a2, false, bf, (short)0, acc2, false, false);
            acc3 = __builtin_amdgcn_wmma_f32_16x16x32_f16(false, a3, false, bf, (short)0, acc3, false, false);
        }

        const float bv   = bias[g * HID + n0 + hrow];
        const int   roff = (lane >> 4) * 8;
        v8f acc[4] = {acc0, acc1, acc2, acc3};
#pragma unroll
        for (int i = 0; i < 4; ++i)
#pragma unroll
            for (int r = 0; r < 8; ++r)
                smem[combo][16 * i + roff + r][hrow] = acc[i][r] + bv;
    }

    __syncthreads();

    // Gate math: 64 x 16 = 1024 elements over 192 threads.
    for (int idx = tid; idx < 64 * 16; idx += 192) {
        int row = idx >> 4, col = idx & 15;
        float ghr = smem[0][row][col], ghz = smem[1][row][col], ghn = smem[2][row][col];
        float gir = smem[3][row][col], giz = smem[4][row][col], gin = smem[5][row][col];
        float r = 1.f / (1.f + __expf(-(gir + ghr)));
        float z = 1.f / (1.f + __expf(-(giz + ghz)));
        float n = tanhf(gin + r * ghn);
        size_t hi = (size_t)(m0 + row) * HID + (n0 + col);
        float  hn = (1.f - z) * n + z * hOldF[hi];
        hNewF[hi] = hn;
        hNewH[hi] = (_Float16)hn;
    }
}

// ---------------------------------------------------------------------------
// x [B,C,HT,WD] f32  ->  seq [T, B, C*WD] f16   (t indexes HT)
// ---------------------------------------------------------------------------
__global__ void pack_x_kernel(const float* __restrict__ x, _Float16* __restrict__ seq) {
    size_t o   = (size_t)blockIdx.x * blockDim.x + threadIdx.x;
    int    t   = (int)(o / ((size_t)BATCH * IN));
    int    rem = (int)(o % ((size_t)BATCH * IN));
    int    b   = rem / IN;
    int    cw  = rem % IN;
    int    c   = cw / WD;
    int    w   = cw % WD;
    size_t xi  = (((size_t)b * CH + c) * HT + t) * WD + w;
    seq[o] = (_Float16)x[xi];
}

__global__ void pack_f16_kernel(const float* __restrict__ in, _Float16* __restrict__ out) {
    size_t i = (size_t)blockIdx.x * blockDim.x + threadIdx.x;
    out[i] = (_Float16)in[i];
}

// h0 [L,HID] broadcast -> hF [L,B,HID] f32 and hH f16 (parity-0 buffers)
__global__ void h_init_kernel(const float* __restrict__ h0,
                              float* __restrict__ hF, _Float16* __restrict__ hH) {
    size_t i = (size_t)blockIdx.x * blockDim.x + threadIdx.x;
    int l = (int)(i / ((size_t)BATCH * HID));
    int j = (int)(i % HID);
    float v = h0[l * HID + j];
    hF[i] = v;
    hH[i] = (_Float16)v;
}

// ---------------------------------------------------------------------------
// Row softmax over DM=512; one block (256 thr) per batch row.
// ---------------------------------------------------------------------------
__global__ void softmax_kernel(const float* __restrict__ logits, float* __restrict__ out) {
    __shared__ float red[256];
    int b = blockIdx.x, tid = threadIdx.x;
    const float* row = logits + (size_t)b * DM;
    float v0 = row[tid], v1 = row[tid + 256];
    red[tid] = fmaxf(v0, v1);
    __syncthreads();
    for (int s = 128; s > 0; s >>= 1) { if (tid < s) red[tid] = fmaxf(red[tid], red[tid + s]); __syncthreads(); }
    float mx = red[0];
    __syncthreads();
    float e0 = __expf(v0 - mx), e1 = __expf(v1 - mx);
    red[tid] = e0 + e1;
    __syncthreads();
    for (int s = 128; s > 0; s >>= 1) { if (tid < s) red[tid] += red[tid + s]; __syncthreads(); }
    float inv = 1.f / red[0];
    out[(size_t)b * DM + tid]       = e0 * inv;
    out[(size_t)b * DM + tid + 256] = e1 * inv;
}

// ---------------------------------------------------------------------------
extern "C" void kernel_launch(void* const* d_in, const int* in_sizes, int n_in,
                              void* d_out, int out_size, void* d_ws, size_t ws_size,
                              hipStream_t stream) {
    const float* x    = (const float*)d_in[0];
    const float* h0   = (const float*)d_in[1];
    const float* Wih[3] = { (const float*)d_in[2], (const float*)d_in[6],  (const float*)d_in[10] };
    const float* Whh[3] = { (const float*)d_in[3], (const float*)d_in[7],  (const float*)d_in[11] };
    const float* bih[3] = { (const float*)d_in[4], (const float*)d_in[8],  (const float*)d_in[12] };
    const float* bhh[3] = { (const float*)d_in[5], (const float*)d_in[9],  (const float*)d_in[13] };
    const float* W1 = (const float*)d_in[14];
    const float* b1 = (const float*)d_in[15];
    const float* W2 = (const float*)d_in[16];
    const float* b2 = (const float*)d_in[17];

    // ---- workspace carve-up (all chunk sizes multiples of 256B)
    char* ws = (char*)d_ws;
    _Float16* seq  = (_Float16*)ws; ws += (size_t)HT * BATCH * IN * 2;        // 77 MB
    _Float16* gx   = (_Float16*)ws; ws += (size_t)HT * BATCH * TH3 * 2;       // 176 MB
    _Float16* Wih0h = (_Float16*)ws; ws += (size_t)TH3 * IN * 2;
    _Float16* WhhH[3]; _Float16* WihH[3];
    WihH[0] = Wih0h;
    for (int l = 0; l < 3; ++l) { WhhH[l] = (_Float16*)ws; ws += (size_t)TH3 * HID * 2; }
    for (int l = 1; l < 3; ++l) { WihH[l] = (_Float16*)ws; ws += (size_t)TH3 * HID * 2; }
    _Float16* W1h = (_Float16*)ws; ws += (size_t)HID * HID * 2;
    _Float16* W2h = (_Float16*)ws; ws += (size_t)DM * HID * 2;
    float*    hFb = (float*)ws;    ws += (size_t)2 * 3 * BATCH * HID * 4;     // double-buffered
    _Float16* hHb = (_Float16*)ws; ws += (size_t)2 * 3 * BATCH * HID * 2;
    _Float16* h1h = (_Float16*)ws; ws += (size_t)BATCH * HID * 2;
    float*    log_ = (float*)ws;   ws += (size_t)BATCH * DM * 4;

    auto HF = [&](int p, int l) { return hFb + ((size_t)p * 3 + l) * BATCH * HID; };
    auto HH = [&](int p, int l) { return hHb + ((size_t)p * 3 + l) * BATCH * HID; };

    // ---- pack inputs/weights to f16
    pack_x_kernel<<<(HT * BATCH * IN) / 256, 256, 0, stream>>>(x, seq);
    pack_f16_kernel<<<(TH3 * IN) / 256, 256, 0, stream>>>(Wih[0], Wih0h);
    for (int l = 0; l < 3; ++l)
        pack_f16_kernel<<<(TH3 * HID) / 256, 256, 0, stream>>>(Whh[l], WhhH[l]);
    for (int l = 1; l < 3; ++l)
        pack_f16_kernel<<<(TH3 * HID) / 256, 256, 0, stream>>>(Wih[l], WihH[l]);
    pack_f16_kernel<<<(HID * HID) / 256, 256, 0, stream>>>(W1, W1h);
    pack_f16_kernel<<<(DM * HID) / 256, 256, 0, stream>>>(W2, W2h);
    h_init_kernel<<<(3 * BATCH * HID) / 256, 256, 0, stream>>>(h0, HF(0, 0), HH(0, 0));

    // ---- layer-0 input projections for ALL timesteps (big parallel GEMM)
    gemm_wmma_kernel<true, false><<<dim3(TH3 / 64, (HT * BATCH) / 64), 128, 0, stream>>>(
        seq, Wih0h, bih[0], gx, HT * BATCH, TH3, IN);

    // ---- recurrence: 224 steps x 3 fused layer-stage kernels
    const dim3 gStage(HID / 16, BATCH / 64);   // (32, 4)
    for (int t = 0; t < HT; ++t) {
        const int pr = t & 1, pw = pr ^ 1;     // read parity, write parity
        gru_stage_kernel<true><<<gStage, 192, 0, stream>>>(
            HH(pr, 0), HF(pr, 0), nullptr, gx + (size_t)t * BATCH * TH3,
            nullptr, WhhH[0], nullptr, bhh[0], HF(pw, 0), HH(pw, 0));
        for (int l = 1; l < 3; ++l)
            gru_stage_kernel<false><<<gStage, 192, 0, stream>>>(
                HH(pr, l), HF(pr, l), HH(pw, l - 1), nullptr,
                WihH[l], WhhH[l], bih[l], bhh[l], HF(pw, l), HH(pw, l));
    }

    // ---- head: relu(h W1^T + b1) W2^T + b2 -> softmax   (final h in parity 0)
    gemm_wmma_kernel<true, true><<<dim3(HID / 64, BATCH / 64), 128, 0, stream>>>(
        HH(0, 2), W1h, b1, h1h, BATCH, HID, HID);
    gemm_wmma_kernel<false, false><<<dim3(DM / 64, BATCH / 64), 128, 0, stream>>>(
        h1h, W2h, b2, log_, BATCH, DM, HID);
    softmax_kernel<<<BATCH, 256, 0, stream>>>(log_, (float*)d_out);
}
